// UpSampler_22247930593943
// MI455X (gfx1250) — matrix operand order repaired and checked by reading, not compiled, not run on hardware
//
#include <hip/hip_runtime.h>

typedef __attribute__((ext_vector_type(16))) _Float16     v16h;
typedef __attribute__((ext_vector_type(8)))  float        v8f;
typedef __attribute__((ext_vector_type(8)))  unsigned int v8u;

#define HH 192
#define WW 192
#define BB 4
#define CC 128       // feature channels
#define CP1 224      // conv1 input channels padded (213 -> 224)
#define HW (HH*WW)

// scheduler pipelining directives (no-op if builtin unavailable)
#if defined(__has_builtin)
#if __has_builtin(__builtin_amdgcn_sched_group_barrier)
#define SGB(mask, size) __builtin_amdgcn_sched_group_barrier((mask), (size), 0)
#endif
#endif
#ifndef SGB
#define SGB(mask, size)
#endif

// ---------------------------------------------------------------------------
// Weight fragment builder: repack (cout, cin, 3, 3) fp32 -> f16 fragments in
// the exact per-lane order the WMMA B operand (32x16, f16) wants.
// Layout: frag[(((nt*9 + tap)*KC + kc)*32 + lane)*16 + h]
//   B element: column n = lane&15, K = kc*32 + (h<8 ? (lane>>4)*8 + h : 16 + (lane>>4)*8 + h-8)
// ---------------------------------------------------------------------------
__global__ void build_wfrag(const float* __restrict__ w, _Float16* __restrict__ frag,
                            int CinR, int CoutR, int KC, int total) {
    int idx = blockIdx.x * blockDim.x + threadIdx.x;
    if (idx >= total) return;
    int h    = idx & 15;
    int t    = idx >> 4;
    int lane = t & 31;  t >>= 5;
    int kc   = t % KC;  t /= KC;
    int tap  = t % 9;
    int nt   = t / 9;
    int k    = kc * 32 + ((lane >> 4) * 8) + (h & 7) + ((h >> 3) * 16);
    int cout = nt * 16 + (lane & 15);
    float v = 0.f;
    if (k < CinR && cout < CoutR)
        v = w[(size_t)(cout * CinR + k) * 9 + tap];
    frag[idx] = (_Float16)v;
}

// ---------------------------------------------------------------------------
// Correlation (9x9, 128-ch dot) + softmax(81) + pack concat input as NHWC f16
// padded to 224 channels:  [attn 0..80 | feature_0 81..208 | flow 209..212 | 0]
// One 16x16 pixel tile per 256-thread block; f1 neighborhood staged in LDS,
// 16 channels at a time (24*24*16 fp32 = 36.9 KB).
// ---------------------------------------------------------------------------
__global__ __launch_bounds__(256)
void corr_softmax_pack(const float* __restrict__ f0, const float* __restrict__ f1,
                       const float* __restrict__ flow0, const float* __restrict__ wflow1,
                       _Float16* __restrict__ x0buf) {
    __shared__ float lds[16 * 576];   // [c 16][row 24][col 24]

    int blk = blockIdx.x;                       // 4 * 12 * 12 = 576 blocks
    int b   = blk / 144;
    int rem = blk % 144;
    int ty0 = (rem / 12) * 16;
    int tx0 = (rem % 12) * 16;
    int tid = threadIdx.x;
    int px  = tid & 15, py = tid >> 4;
    int x = tx0 + px, y = ty0 + py;

    float acc[81];
#pragma unroll
    for (int i = 0; i < 81; ++i) acc[i] = 0.f;

    const float* f0p = f0 + (size_t)b * CC * HW + (size_t)y * WW + x;
    _Float16* xo = x0buf + (size_t)(((b * HH + y) * WW) + x) * CP1;

#pragma unroll 1
    for (int cc = 0; cc < 8; ++cc) {
        int c0 = cc * 16;
        __syncthreads();
        // cooperative load of f1 tile: rows ty0-4..ty0+19, cols tx0-4..tx0+19
        for (int i = tid; i < 16 * 576; i += 256) {
            int c  = i / 576;
            int r  = i % 576;
            int ry = r / 24, rx = r % 24;
            int gy = ty0 - 4 + ry, gx = tx0 - 4 + rx;
            float v = 0.f;
            if (gy >= 0 && gy < HH && gx >= 0 && gx < WW)
                v = f1[(size_t)(b * CC + c0 + c) * HW + (size_t)gy * WW + gx];
            lds[i] = v;
        }
        __syncthreads();

        float f0r[16];
#pragma unroll
        for (int c = 0; c < 16; ++c) f0r[c] = f0p[(size_t)(c0 + c) * HW];

#pragma unroll 1
        for (int c = 0; c < 16; ++c) {
            float a = f0r[c];
            const float* row = lds + c * 576;
#pragma unroll
            for (int dy = 0; dy < 9; ++dy) {
                const float* rr = row + (py + dy) * 24 + px;
#pragma unroll
                for (int dx = 0; dx < 9; ++dx)
                    acc[dy * 9 + dx] = fmaf(a, rr[dx], acc[dy * 9 + dx]);
            }
        }
        // pack this channel chunk of feature_0 into the concat buffer
#pragma unroll
        for (int c = 0; c < 16; ++c) xo[81 + c0 + c] = (_Float16)f0r[c];
    }

    // softmax over the 81 correlation values
    float m = acc[0];
#pragma unroll
    for (int i = 1; i < 81; ++i) m = fmaxf(m, acc[i]);
    float s = 0.f;
#pragma unroll
    for (int i = 0; i < 81; ++i) { acc[i] = __expf(acc[i] - m); s += acc[i]; }
    float inv = 1.0f / s;
#pragma unroll
    for (int i = 0; i < 81; ++i) xo[i] = (_Float16)(acc[i] * inv);

    size_t pofs = (size_t)y * WW + x;
    xo[209] = (_Float16)flow0 [(size_t)(b * 2 + 0) * HW + pofs];
    xo[210] = (_Float16)flow0 [(size_t)(b * 2 + 1) * HW + pofs];
    xo[211] = (_Float16)wflow1[(size_t)(b * 2 + 0) * HW + pofs];
    xo[212] = (_Float16)wflow1[(size_t)(b * 2 + 1) * HW + pofs];
#pragma unroll
    for (int c = 213; c < 224; ++c) xo[c] = (_Float16)0.f;
}

// ---------------------------------------------------------------------------
// Implicit-GEMM 3x3 conv (pad=1) via v_wmma_f32_16x16x32_f16.
// One wave per 32-pixel x-strip (two 16-pixel M tiles sharing B fragments).
// Branch-free A loads (clamped address + per-lane zero select) keep each tap
// a single scheduling region; sched_group_barrier builds a depth-2 software
// pipeline: loads of K-chunk kc+1 issue while WMMAs of chunk kc execute.
// ---------------------------------------------------------------------------
template<int NT, int KC, bool LRELU, bool FINAL>
__global__ __launch_bounds__(256)
void conv3x3_wmma(const _Float16* __restrict__ in, const _Float16* __restrict__ wfrag,
                  const float* __restrict__ bias, _Float16* __restrict__ out,
                  float* __restrict__ fout) {
    constexpr int CinP  = KC * 32;
    constexpr int CoutP = NT * 16;
    constexpr int LOADS = 4 + 2 * NT;   // b128 loads per K-chunk (A + B)
    constexpr int MMAS  = 2 * NT;       // WMMAs per K-chunk

    int wave = (blockIdx.x * blockDim.x + threadIdx.x) >> 5;   // 4608 waves
    int lane = threadIdx.x & 31;
    int b    = wave / (HH * 6);
    int rem  = wave % (HH * 6);
    int y    = rem / 6;
    int x0t  = (rem % 6) * 32;
    int p    = lane & 15;
    int hsel = lane >> 4;

    v8f acc0[NT] = {};
    v8f acc1[NT] = {};
    const v8u* wf = (const v8u*)wfrag;

#pragma unroll 1
    for (int tap = 0; tap < 9; ++tap) {
        int dy = tap / 3 - 1;
        int dx = tap % 3 - 1;
        int ys = y + dy;
        if (ys < 0 || ys >= HH) continue;            // zero-pad row (wave-uniform)
        int xs0 = x0t + p + dx;                      // tile-0 pixel
        int xs1 = xs0 + 16;                          // tile-1 pixel
        bool inb0 = (xs0 >= 0);                      // xs0 < WW always (x0t <= 160)
        bool inb1 = (xs1 < WW);                      // xs1 >= 15 always
        int xc0 = inb0 ? xs0 : 0;                    // clamped: always valid memory
        int xc1 = inb1 ? xs1 : (WW - 1);
        const _Float16* rowb = in + (size_t)((b * HH + ys) * WW) * CinP;
        const _Float16* row0 = rowb + (size_t)xc0 * CinP;
        const _Float16* row1 = rowb + (size_t)xc1 * CinP;

#pragma unroll
        for (int kc = 0; kc < KC; ++kc) {
            // unconditional A loads from clamped addresses
            uint4 lo0 = *(const uint4*)(row0 + kc * 32 + hsel * 8);
            uint4 hi0 = *(const uint4*)(row0 + kc * 32 + hsel * 8 + 16);
            uint4 lo1 = *(const uint4*)(row1 + kc * 32 + hsel * 8);
            uint4 hi1 = *(const uint4*)(row1 + kc * 32 + hsel * 8 + 16);
            // B fragments batched into distinct registers
            v8u bu[NT];
#pragma unroll
            for (int n = 0; n < NT; ++n)
                bu[n] = wf[(size_t)(((n * 9 + tap) * KC + kc) * 32 + lane)];

            // zero OOB lanes with selects (no exec branches)
            v8u au0, au1;
            au0[0] = inb0 ? lo0.x : 0u; au0[1] = inb0 ? lo0.y : 0u;
            au0[2] = inb0 ? lo0.z : 0u; au0[3] = inb0 ? lo0.w : 0u;
            au0[4] = inb0 ? hi0.x : 0u; au0[5] = inb0 ? hi0.y : 0u;
            au0[6] = inb0 ? hi0.z : 0u; au0[7] = inb0 ? hi0.w : 0u;
            au1[0] = inb1 ? lo1.x : 0u; au1[1] = inb1 ? lo1.y : 0u;
            au1[2] = inb1 ? lo1.z : 0u; au1[3] = inb1 ? lo1.w : 0u;
            au1[4] = inb1 ? hi1.x : 0u; au1[5] = inb1 ? hi1.y : 0u;
            au1[6] = inb1 ? hi1.z : 0u; au1[7] = inb1 ? hi1.w : 0u;

            v16h a0 = __builtin_bit_cast(v16h, au0);
            v16h a1 = __builtin_bit_cast(v16h, au1);
#pragma unroll
            for (int n = 0; n < NT; ++n) {
                v16h bm = __builtin_bit_cast(v16h, bu[n]);
                acc0[n] = __builtin_amdgcn_wmma_f32_16x16x32_f16(
                    false, a0, false, bm, (short)0, acc0[n], false, false);
                acc1[n] = __builtin_amdgcn_wmma_f32_16x16x32_f16(
                    false, a1, false, bm, (short)0, acc1[n], false, false);
            }
        }

        // software pipeline: [L kc0][L kc1][W kc0][L kc2][W kc1]...[W last]
        SGB(0x020, LOADS);
#pragma unroll
        for (int kc = 1; kc < KC; ++kc) {
            SGB(0x020, LOADS);
            SGB(0x008, MMAS);
        }
        SGB(0x008, MMAS);
    }

    // epilogue: D layout -> M = r + 8*(lane>>4), N (cout) = lane&15
    if (!FINAL) {
#pragma unroll
        for (int n = 0; n < NT; ++n) {
            int cout = n * 16 + p;
            float bv = bias[cout];
#pragma unroll
            for (int r = 0; r < 8; ++r) {
                int m  = r + 8 * hsel;
                float v0 = acc0[n][r] + bv;
                float v1 = acc1[n][r] + bv;
                if (LRELU) {
                    v0 = (v0 > 0.f) ? v0 : 0.1f * v0;
                    v1 = (v1 > 0.f) ? v1 : 0.1f * v1;
                }
                size_t rowo = (size_t)((b * HH + y) * WW);
                out[(rowo + x0t + m)      * CoutP + cout] = (_Float16)v0;
                out[(rowo + x0t + 16 + m) * CoutP + cout] = (_Float16)v1;
            }
        }
    } else {
        // conv7: only cout 0..1 are real; write fp32 NCHW to d_out
        if (p < 2) {
            float bv = bias[p];
            size_t rowo = (size_t)(((b * 2 + p) * HH + y) * WW);
#pragma unroll
            for (int r = 0; r < 8; ++r) {
                int m = r + 8 * hsel;
                fout[rowo + x0t + m]      = acc0[0][r] + bv;
                fout[rowo + x0t + 16 + m] = acc1[0][r] + bv;
            }
        }
    }
}

// ---------------------------------------------------------------------------
// Orchestration
// ---------------------------------------------------------------------------
extern "C" void kernel_launch(void* const* d_in, const int* in_sizes, int n_in,
                              void* d_out, int out_size, void* d_ws, size_t ws_size,
                              hipStream_t stream) {
    (void)in_sizes; (void)n_in; (void)out_size; (void)ws_size;

    const float* f0     = (const float*)d_in[0];
    const float* f1     = (const float*)d_in[1];
    const float* flow0  = (const float*)d_in[2];
    const float* wflow1 = (const float*)d_in[3];
    const float* w[7];
    const float* bs[7];
    for (int i = 0; i < 7; ++i) {
        w[i]  = (const float*)d_in[4 + 2 * i];
        bs[i] = (const float*)d_in[5 + 2 * i];
    }

    // workspace layout (bytes)
    constexpr size_t NPIX      = (size_t)BB * HH * WW;                 // 147456
    constexpr size_t X0_BYTES  = NPIX * CP1 * 2;                       // 66.1 MB
    constexpr size_t ACT_BYTES = NPIX * 96 * 2;                        // 28.3 MB
    // conv shapes: {CinR, CoutR, KC, NT}
    constexpr int CinR [7] = {213, 96, 96, 96, 96, 64, 32};
    constexpr int CoutR[7] = { 96, 96, 96, 96, 64, 32,  2};
    constexpr int KC   [7] = {  7,  3,  3,  3,  3,  2,  1};
    constexpr int NT   [7] = {  6,  6,  6,  6,  4,  2,  1};

    char* ws = (char*)d_ws;
    _Float16* x0   = (_Float16*)ws;
    _Float16* actA = (_Float16*)(ws + X0_BYTES);
    _Float16* actB = (_Float16*)(ws + X0_BYTES + ACT_BYTES);
    size_t off = X0_BYTES + 2 * ACT_BYTES;
    _Float16* frag[7];
    for (int i = 0; i < 7; ++i) {
        frag[i] = (_Float16*)(ws + off);
        off += (size_t)NT[i] * 9 * KC[i] * 512 * 2;
    }

    // 1) build weight fragments
    for (int i = 0; i < 7; ++i) {
        int total = NT[i] * 9 * KC[i] * 512;
        build_wfrag<<<(total + 255) / 256, 256, 0, stream>>>(
            w[i], frag[i], CinR[i], CoutR[i], KC[i], total);
    }

    // 2) correlation + softmax + concat pack
    corr_softmax_pack<<<BB * 12 * 12, 256, 0, stream>>>(f0, f1, flow0, wflow1, x0);

    // 3) conv stack: 4608 waves (32-pixel strips) -> 8 waves/block -> 576 blocks
    const int CBLK = (BB * HH * 6) / 8;   // 576
    conv3x3_wmma<6, 7, true,  false><<<CBLK, 256, 0, stream>>>(x0,   frag[0], bs[0], actA, nullptr);
    conv3x3_wmma<6, 3, true,  false><<<CBLK, 256, 0, stream>>>(actA, frag[1], bs[1], actB, nullptr);
    conv3x3_wmma<6, 3, true,  false><<<CBLK, 256, 0, stream>>>(actB, frag[2], bs[2], actA, nullptr);
    conv3x3_wmma<6, 3, true,  false><<<CBLK, 256, 0, stream>>>(actA, frag[3], bs[3], actB, nullptr);
    conv3x3_wmma<4, 3, true,  false><<<CBLK, 256, 0, stream>>>(actB, frag[4], bs[4], actA, nullptr);
    conv3x3_wmma<2, 2, true,  false><<<CBLK, 256, 0, stream>>>(actA, frag[5], bs[5], actB, nullptr);
    conv3x3_wmma<1, 1, false, true ><<<CBLK, 256, 0, stream>>>(actB, frag[6], bs[6], nullptr,
                                                               (float*)d_out);
}
